// DynamicGNN_8478265442579
// MI455X (gfx1250) — compile-verified, matrix operand order checked by reading, not run on hardware
//
#include <hip/hip_runtime.h>
#include <hip/hip_bf16.h>

// DGCNN forward on gfx1250 (MI455X), wave32 + WMMA f32_16x16x32_f16.
// Pipeline: pack f16 features -> per layer {sqnorm, WMMA-kNN+topK, fused
// WMMA EdgeConv (gather + MLP + K-max)} -> WMMA proj GEMMs -> segment max
// -> tiny head MLP + log_softmax.

#define BGRAPH 16
#define NPTS   2048
#define TOT    (BGRAPH * NPTS)   // 32768
#define HID    64
#define KNN    20
#define NCLS   40

typedef _Float16 h8   __attribute__((ext_vector_type(8)));
typedef _Float16 v16h __attribute__((ext_vector_type(16)));
typedef float    v8f  __attribute__((ext_vector_type(8)));
typedef float    f8v  __attribute__((ext_vector_type(8)));

// ---- WMMA fragment helpers (wave32 layouts per CDNA5 ISA 7.12.2) ----------
__device__ __forceinline__ v16h mk16(h8 lo, h8 hi) {
  v16h r;
#pragma unroll
  for (int i = 0; i < 8; ++i) { r[i] = lo[i]; r[i + 8] = hi[i]; }
  return r;
}

// A (16xK rows) / B (as Bt[n][k]) fragment from row-major f16 storage, row
// stride S halves.  Lane l: row = l&15, k-chunks at kc*32 + {kb, kb+16},
// kb = (l>>4)*8.
template <int S>
__device__ __forceinline__ v16h load_frag(const _Float16* base, int lane, int kc) {
  const int row = lane & 15;
  const int kb  = (lane >> 4) << 3;
  const _Float16* p = base + row * S + kc * 32 + kb;
  return mk16(*(const h8*)p, *(const h8*)(p + 16));
}

// A fragment from f32 storage (converted to f16 in-register).
template <int S>
__device__ __forceinline__ v16h load_frag_f32(const float* base, int lane, int kc) {
  const int row = lane & 15;
  const int kb  = (lane >> 4) << 3;
  const float* p = base + (size_t)row * S + kc * 32 + kb;
  f8v lo = *(const f8v*)p;
  f8v hi = *(const f8v*)(p + 16);
  v16h r;
#pragma unroll
  for (int i = 0; i < 8; ++i) { r[i] = (_Float16)lo[i]; r[i + 8] = (_Float16)hi[i]; }
  return r;
}

// Edge-feature fragment: logical row e = [ xi(0..HP) | xj-xi(0..HP) ].
// HP is a multiple of 16, so each 8-half chunk lies fully in one half and the
// half-select folds to a compile-time constant per unrolled kc.
template <int HP>
__device__ __forceinline__ v16h load_edge_frag(const _Float16* xi, const _Float16* xj,
                                               int lane, int kc) {
  const int kb = (lane >> 4) << 3;
  const int o0 = kc * 32 + kb;
  const int o1 = o0 + 16;
  h8 lo = ((kc * 32) < HP)
              ? *(const h8*)(xi + o0)
              : (*(const h8*)(xj + (o0 - HP)) - *(const h8*)(xi + (o0 - HP)));
  h8 hi = ((kc * 32 + 16) < HP)
              ? *(const h8*)(xi + o1)
              : (*(const h8*)(xj + (o1 - HP)) - *(const h8*)(xi + (o1 - HP)));
  return mk16(lo, hi);
}

__device__ __forceinline__ float elu1(float v) {
  return v > 0.f ? v : (expf(v) - 1.f);
}

// float atomic max via int/uint punning (valid with -inf init).
__device__ __forceinline__ void atomicMaxF32(float* a, float v) {
  if (v >= 0.f) atomicMax((int*)a, __float_as_int(v));
  else          atomicMin((unsigned int*)a, (unsigned int)__float_as_int(v));
}

// ---- small packing kernels -------------------------------------------------
__global__ __launch_bounds__(256) void pack_x_kernel(const float* __restrict__ x,
                                                     _Float16* __restrict__ xh) {
  int i = blockIdx.x * 256 + threadIdx.x;
  if (i >= TOT) return;
  _Float16* r = xh + (size_t)i * HID;
  float v0 = x[i * 3 + 0], v1 = x[i * 3 + 1], v2 = x[i * 3 + 2];
  r[0] = (_Float16)v0; r[1] = (_Float16)v1; r[2] = (_Float16)v2;
#pragma unroll
  for (int c = 3; c < HID; ++c) r[c] = (_Float16)0.f;
}

__global__ __launch_bounds__(256) void sqn_kernel(const _Float16* __restrict__ xh,
                                                  float* __restrict__ sqn) {
  int i = blockIdx.x * 256 + threadIdx.x;
  if (i >= TOT) return;
  const _Float16* r = xh + (size_t)i * HID;
  float s = 0.f;
#pragma unroll
  for (int c = 0; c < HID; ++c) { float v = (float)r[c]; s += v * v; }
  sqn[i] = s;
}

// dst f16 [64][KP] = transpose of src f32 [KIN][64], zero padded.
// special: layer-0 EdgeConv W1 ([6][64]) split-mapped to e-pad layout
// (k<3 -> row k ; 16<=k<19 -> row k-13 ; else 0).
__global__ __launch_bounds__(256) void packWt_kernel(_Float16* __restrict__ dst,
                                                     const float* __restrict__ src,
                                                     int KIN, int KP, int special) {
  int i = blockIdx.x * 256 + threadIdx.x;
  if (i >= 64 * KP) return;
  int n = i / KP, k = i % KP;
  int sk;
  if (special) sk = (k < 3) ? k : ((k >= 16 && k < 19) ? (k - 13) : -1);
  else         sk = (k < KIN) ? k : -1;
  dst[n * KP + k] = (sk >= 0) ? (_Float16)src[sk * 64 + n] : (_Float16)0.f;
}

// ---- kNN: WMMA Gram tiles + register top-K ---------------------------------
// One wave per 16-row strip; 128 column tiles of 16; rank by sqn[j] - 2*<i,j>
// (row-constant |xi|^2 dropped; self is provably rank-1 -> matches reference's
// self-inclusive kNN).  All 32 lanes select: lane s scans cols 0..7 of row s,
// lane s+16 scans cols 8..15; per-lane top-20 lists are merged at the end.
__global__ __launch_bounds__(128) void knn_kernel(const _Float16* __restrict__ xh,
                                                  const float* __restrict__ sqn,
                                                  int* __restrict__ nbr) {
  __shared__ __align__(16) float tile[4][16][17];
  __shared__ float sd[4][32][KNN];
  __shared__ int   si[4][32][KNN];
  const int wave = threadIdx.x >> 5, lane = threadIdx.x & 31;
  const int strip = blockIdx.x * 4 + wave;
  const int b  = strip >> 7;                  // / (NPTS/16)
  const int r0 = (strip & 127) << 4;
  const int gbase = b * NPTS;
  const _Float16* rowbase = xh + (size_t)(gbase + r0) * HID;

  // row A-fragments are loop-invariant
  const v16h a0 = load_frag<HID>(rowbase, lane, 0);
  const v16h a1 = load_frag<HID>(rowbase, lane, 1);

  float bd[KNN]; int bi[KNN];
#pragma unroll
  for (int q = 0; q < KNN; ++q) { bd[q] = 3.4e38f; bi[q] = 0; }

  const int moff = (lane < 16) ? 0 : 8;
  const int row  = lane & 15;                 // also the column n this lane owns
  const int jjb  = (lane >> 4) << 3;          // candidate column sub-range 0 / 8
  for (int t = 0; t < NPTS / 16; ++t) {
    const float sc = sqn[gbase + t * 16 + row];
    const _Float16* colbase = xh + (size_t)(gbase + t * 16) * HID;
    v8f acc = {};
    v16h b0 = load_frag<HID>(colbase, lane, 0);
    acc = __builtin_amdgcn_wmma_f32_16x16x32_f16(false, a0, false, b0, (short)0, acc, false, false);
    v16h b1 = load_frag<HID>(colbase, lane, 1);
    acc = __builtin_amdgcn_wmma_f32_16x16x32_f16(false, a1, false, b1, (short)0, acc, false, false);
#pragma unroll
    for (int r = 0; r < 8; ++r)
      tile[wave][moff + r][row] = sc - 2.f * acc[r];   // d2 - |xi|^2

    // batch the 8 candidates for this lane into registers (one DS wait)
    float cand[8];
#pragma unroll
    for (int u = 0; u < 8; ++u) cand[u] = tile[wave][row][jjb + u];
#pragma unroll
    for (int u = 0; u < 8; ++u) {
      float d = cand[u];
      if (d < bd[KNN - 1]) {
        float cd = d; int ci = t * 16 + jjb + u;
#pragma unroll
        for (int q = 0; q < KNN; ++q) {
          if (cd < bd[q]) {
            float td = bd[q]; int ti = bi[q];
            bd[q] = cd; bi[q] = ci; cd = td; ci = ti;
          }
        }
      }
    }
  }

  // dump per-lane sorted lists, then lanes 0..15 merge their pair (DS is
  // in-order within a wave; lists are wave-private -> no barrier needed)
#pragma unroll
  for (int q = 0; q < KNN; ++q) { sd[wave][lane][q] = bd[q]; si[wave][lane][q] = bi[q]; }
  if (lane < 16) {
    int ia = 0, ib = 0;
    const size_t gp = (size_t)(gbase + r0 + lane);
    for (int q = 0; q < KNN; ++q) {
      float da = sd[wave][lane][ia];
      float db = sd[wave][lane + 16][ib];
      bool ta = (da <= db);
      int ci = ta ? si[wave][lane][ia] : si[wave][lane + 16][ib];
      nbr[gp * KNN + q] = ci;
      ia += ta ? 1 : 0;
      ib += ta ? 0 : 1;
    }
  }
}

// ---- fused EdgeConv: gather -> GEMM1 -> ELU -> GEMM2 -> max over K ---------
// Workgroup: 16 points (= 320 edges = 20 M-tiles), 20 waves, 1 M-tile/wave.
template <int HP, int EKP>
__global__ __launch_bounds__(640)
void edgeconv_kernel(const _Float16* __restrict__ xh, const int* __restrict__ nbr,
                     const _Float16* __restrict__ W1t, const float* __restrict__ b1,
                     const _Float16* __restrict__ W2t, const float* __restrict__ b2,
                     float* __restrict__ xs, int loff,
                     _Float16* __restrict__ xh_next) {
  __shared__ __align__(16) _Float16 h1lds[20][16][HID];
  __shared__ __align__(16) float outLds[16][HID];
  const int tid = threadIdx.x;
  const int wave = tid >> 5, lane = tid & 31;
  const int b  = blockIdx.x / (NPTS / 16);
  const int p0 = (blockIdx.x % (NPTS / 16)) * 16;
  const int gbase = b * NPTS;

  for (int i = tid; i < 16 * HID; i += 640) ((float*)outLds)[i] = -__builtin_inff();
  __syncthreads();

  // per-lane edge -> (point, neighbor) feature pointers
  const int erow = wave * 16 + (lane & 15);
  const int p    = p0 + erow / KNN;
  const int kk   = erow % KNN;
  const int j    = nbr[(size_t)(gbase + p) * KNN + kk];
  const _Float16* xi = xh + (size_t)(gbase + p) * HID;
  const _Float16* xj = xh + (size_t)(gbase + j) * HID;

  // GEMM1: 16 x EKP @ EKP x 64
  v8f acc[4] = {};
#pragma unroll
  for (int kc = 0; kc < EKP / 32; ++kc) {
    const v16h a = load_edge_frag<HP>(xi, xj, lane, kc);
#pragma unroll
    for (int nt = 0; nt < 4; ++nt) {
      v16h bf = load_frag<EKP>(W1t + nt * 16 * EKP, lane, kc);
      acc[nt] = __builtin_amdgcn_wmma_f32_16x16x32_f16(false, a, false, bf, (short)0, acc[nt], false, false);
    }
  }
  // bias + ELU -> LDS (C layout -> row-major re-stage for next A operand)
  {
    const int mb = (lane < 16) ? 0 : 8;
#pragma unroll
    for (int nt = 0; nt < 4; ++nt) {
      const int n = (lane & 15) + nt * 16;
      const float bb = b1[n];
#pragma unroll
      for (int r = 0; r < 8; ++r)
        h1lds[wave][mb + r][n] = (_Float16)elu1(acc[nt][r] + bb);
    }
  }
  // GEMM2: 16 x 64 @ 64 x 64
  v8f acc2[4] = {};
#pragma unroll
  for (int kc = 0; kc < 2; ++kc) {
    const v16h a = load_frag<HID>(&h1lds[wave][0][0], lane, kc);
#pragma unroll
    for (int nt = 0; nt < 4; ++nt) {
      v16h bf = load_frag<HID>(W2t + nt * 16 * HID, lane, kc);
      acc2[nt] = __builtin_amdgcn_wmma_f32_16x16x32_f16(false, a, false, bf, (short)0, acc2[nt], false, false);
    }
  }
  // bias + max over the K edges of each point
  {
    const int mb = (lane < 16) ? 0 : 8;
#pragma unroll
    for (int nt = 0; nt < 4; ++nt) {
      const int n = (lane & 15) + nt * 16;
      const float bb = b2[n];
#pragma unroll
      for (int r = 0; r < 8; ++r) {
        const int e2 = wave * 16 + mb + r;
        atomicMaxF32(&outLds[e2 / KNN][n], acc2[nt][r] + bb);
      }
    }
  }
  __syncthreads();
  for (int i = tid; i < 16 * HID; i += 640) {
    int pl = i >> 6, n = i & 63;
    float v = outLds[pl][n];
    size_t gp = (size_t)(gbase + p0 + pl);
    xs[gp * (3 * HID) + loff + n] = v;
    xh_next[gp * HID + n] = (_Float16)v;
  }
}

// ---- generic GEMM + bias + ELU: out[M][64] = act(A[M][KD] @ W + b) --------
template <int KD>
__global__ __launch_bounds__(256)
void gemm_bias_act_kernel(const float* __restrict__ A, const _Float16* __restrict__ Wt,
                          const float* __restrict__ bias, float* __restrict__ outp,
                          int M, int doElu) {
  const int wave = threadIdx.x >> 5, lane = threadIdx.x & 31;
  const int r0 = (blockIdx.x * 8 + wave) * 16;
  if (r0 >= M) return;
  const float* abase = A + (size_t)r0 * KD;
  v8f acc[4] = {};
#pragma unroll
  for (int kc = 0; kc < KD / 32; ++kc) {
    const v16h a = load_frag_f32<KD>(abase, lane, kc);
#pragma unroll
    for (int nt = 0; nt < 4; ++nt) {
      v16h bf = load_frag<KD>(Wt + nt * 16 * KD, lane, kc);
      acc[nt] = __builtin_amdgcn_wmma_f32_16x16x32_f16(false, a, false, bf, (short)0, acc[nt], false, false);
    }
  }
  const int mb = (lane < 16) ? 0 : 8;
#pragma unroll
  for (int nt = 0; nt < 4; ++nt) {
    const int n = (lane & 15) + nt * 16;
    const float bb = bias[n];
#pragma unroll
    for (int r = 0; r < 8; ++r) {
      float v = acc[nt][r] + bb;
      if (doElu) v = elu1(v);
      outp[(size_t)(r0 + mb + r) * HID + n] = v;
    }
  }
}

// ---- global max pool per graph --------------------------------------------
__global__ __launch_bounds__(64) void segmax_kernel(const float* __restrict__ h,
                                                    float* __restrict__ g) {
  const int b = blockIdx.x, c = threadIdx.x;
  const float* p = h + (size_t)b * NPTS * HID + c;
  float m = -__builtin_inff();
#pragma unroll 8
  for (int i = 0; i < NPTS; ++i) m = fmaxf(m, p[(size_t)i * HID]);
  g[b * HID + c] = m;
}

// ---- head MLP + log_softmax (tiny) ----------------------------------------
__global__ __launch_bounds__(256)
void head_kernel(const float* __restrict__ g,
                 const float* __restrict__ W1, const float* __restrict__ b1,
                 const float* __restrict__ W2, const float* __restrict__ b2,
                 const float* __restrict__ W3, const float* __restrict__ b3,
                 float* __restrict__ outp) {
  __shared__ float g0[BGRAPH][HID], g1[BGRAPH][HID], g2[BGRAPH][HID], o[BGRAPH][NCLS];
  const int tid = threadIdx.x;
  for (int i = tid; i < BGRAPH * HID; i += 256) ((float*)g0)[i] = g[i];
  __syncthreads();
  for (int i = tid; i < BGRAPH * HID; i += 256) {
    int bb = i >> 6, hh = i & 63;
    float s = b1[hh];
    for (int k = 0; k < HID; ++k) s += g0[bb][k] * W1[k * HID + hh];
    ((float*)g1)[i] = elu1(s);
  }
  __syncthreads();
  for (int i = tid; i < BGRAPH * HID; i += 256) {
    int bb = i >> 6, hh = i & 63;
    float s = b2[hh];
    for (int k = 0; k < HID; ++k) s += g1[bb][k] * W2[k * HID + hh];
    ((float*)g2)[i] = elu1(s);
  }
  __syncthreads();
  for (int i = tid; i < BGRAPH * NCLS; i += 256) {
    int bb = i / NCLS, c = i % NCLS;
    float s = b3[c];
    for (int k = 0; k < HID; ++k) s += g2[bb][k] * W3[k * NCLS + c];
    o[bb][c] = s;
  }
  __syncthreads();
  if (tid < BGRAPH) {
    float mx = -__builtin_inff();
    for (int c = 0; c < NCLS; ++c) mx = fmaxf(mx, o[tid][c]);
    float se = 0.f;
    for (int c = 0; c < NCLS; ++c) se += expf(o[tid][c] - mx);
    float ls = logf(se);
    for (int c = 0; c < NCLS; ++c) outp[tid * NCLS + c] = o[tid][c] - mx - ls;
  }
}

// ---- host ------------------------------------------------------------------
extern "C" void kernel_launch(void* const* d_in, const int* in_sizes, int n_in,
                              void* d_out, int out_size, void* d_ws, size_t ws_size,
                              hipStream_t stream) {
  const float* x = (const float*)d_in[0];
  // d_in[1] = batch ids (implicit in layout, unused)
  const float* cW1[3] = {(const float*)d_in[2], (const float*)d_in[6],  (const float*)d_in[10]};
  const float* cb1[3] = {(const float*)d_in[3], (const float*)d_in[7],  (const float*)d_in[11]};
  const float* cW2[3] = {(const float*)d_in[4], (const float*)d_in[8],  (const float*)d_in[12]};
  const float* cb2[3] = {(const float*)d_in[5], (const float*)d_in[9],  (const float*)d_in[13]};
  const float* pW1 = (const float*)d_in[14]; const float* pb1 = (const float*)d_in[15];
  const float* pW2 = (const float*)d_in[16]; const float* pb2 = (const float*)d_in[17];
  const float* mW1 = (const float*)d_in[18]; const float* mb1 = (const float*)d_in[19];
  const float* mW2 = (const float*)d_in[20]; const float* mb2 = (const float*)d_in[21];
  const float* mW3 = (const float*)d_in[22]; const float* mb3 = (const float*)d_in[23];

  // workspace carve-out (~52 MB)
  char* ws = (char*)d_ws;
  size_t off = 0;
  auto take = [&](size_t bytes) -> char* {
    char* r = ws + off;
    off = (off + bytes + 255) & ~(size_t)255;
    return r;
  };
  _Float16* xhA = (_Float16*)take((size_t)TOT * HID * 2);
  _Float16* xhB = (_Float16*)take((size_t)TOT * HID * 2);
  float*    sqn = (float*)take((size_t)TOT * 4);
  int*      nbr = (int*)take((size_t)TOT * KNN * 4);
  float*    xs  = (float*)take((size_t)TOT * 3 * HID * 4);
  float*    hb1 = (float*)take((size_t)TOT * HID * 4);
  float*    hb2 = (float*)take((size_t)TOT * HID * 4);
  float*    gbuf = (float*)take((size_t)BGRAPH * HID * 4);
  _Float16* W1t[3];
  W1t[0] = (_Float16*)take(64 * 32 * 2);
  W1t[1] = (_Float16*)take(64 * 128 * 2);
  W1t[2] = (_Float16*)take(64 * 128 * 2);
  _Float16* W2t[3];
  for (int l = 0; l < 3; ++l) W2t[l] = (_Float16*)take(64 * 64 * 2);
  _Float16* P1t = (_Float16*)take(64 * 192 * 2);
  _Float16* P2t = (_Float16*)take(64 * 64 * 2);
  (void)in_sizes; (void)n_in; (void)out_size; (void)ws_size;

  pack_x_kernel<<<TOT / 256, 256, 0, stream>>>(x, xhA);
  packWt_kernel<<<(64 * 32  + 255) / 256, 256, 0, stream>>>(W1t[0], cW1[0], 6,   32,  1);
  packWt_kernel<<<(64 * 128 + 255) / 256, 256, 0, stream>>>(W1t[1], cW1[1], 128, 128, 0);
  packWt_kernel<<<(64 * 128 + 255) / 256, 256, 0, stream>>>(W1t[2], cW1[2], 128, 128, 0);
  for (int l = 0; l < 3; ++l)
    packWt_kernel<<<(64 * 64 + 255) / 256, 256, 0, stream>>>(W2t[l], cW2[l], 64, 64, 0);
  packWt_kernel<<<(64 * 192 + 255) / 256, 256, 0, stream>>>(P1t, pW1, 192, 192, 0);
  packWt_kernel<<<(64 * 64  + 255) / 256, 256, 0, stream>>>(P2t, pW2, 64,  64,  0);

  _Float16* cur = xhA;
  _Float16* nxt = xhB;
  for (int l = 0; l < 3; ++l) {
    sqn_kernel<<<TOT / 256, 256, 0, stream>>>(cur, sqn);
    knn_kernel<<<(TOT / 16) / 4, 128, 0, stream>>>(cur, sqn, nbr);
    if (l == 0)
      edgeconv_kernel<16, 32><<<TOT / 16, 640, 0, stream>>>(
          cur, nbr, W1t[0], cb1[0], W2t[0], cb2[0], xs, 0, nxt);
    else if (l == 1)
      edgeconv_kernel<64, 128><<<TOT / 16, 640, 0, stream>>>(
          cur, nbr, W1t[1], cb1[1], W2t[1], cb2[1], xs, 64, nxt);
    else
      edgeconv_kernel<64, 128><<<TOT / 16, 640, 0, stream>>>(
          cur, nbr, W1t[2], cb1[2], W2t[2], cb2[2], xs, 128, nxt);
    _Float16* t = cur; cur = nxt; nxt = t;
  }

  gemm_bias_act_kernel<192><<<TOT / 128, 256, 0, stream>>>(xs,  P1t, pb1, hb1, TOT, 1);
  gemm_bias_act_kernel<64> <<<TOT / 128, 256, 0, stream>>>(hb1, P2t, pb2, hb2, TOT, 1);
  segmax_kernel<<<BGRAPH, 64, 0, stream>>>(hb2, gbuf);
  head_kernel<<<1, 256, 0, stream>>>(gbuf, mW1, mb1, mW2, mb2, mW3, mb3, (float*)d_out);
}
